// SLOG_N_87540023427788
// MI455X (gfx1250) — matrix operand order repaired and checked by reading, not compiled, not run on hardware
//
#include <hip/hip_runtime.h>
#include <hip/hip_bf16.h>
#include <math.h>

// ---------------------------------------------------------------------------
// SLOG_N on MI455X (gfx1250). Compute-bound (working set < 32MB, fits in the
// 192MB L2): all large matmuls run on wave32 WMMA (v_wmma_f32_16x16x32_f16 /
// _bf16, f32 accum). (U*w)@U.T@X is reassociated to (U*w) @ (U^T X).
// Expert kernel tiles 16 adjacent U-columns per block so every global load of
// U is a contiguous 64B run (2x line overfetch instead of 32x column gather).
// ---------------------------------------------------------------------------

#define NN    2000
#define IN_C  128
#define HIDD  256
#define OUTC  64
#define NE    5
#define KPE   400     // N / NE
#define EH    64
#define NPART 8       // n-partitions for the expert kernel (2000 = 8 * 250)
#define BN_EPS 1e-5f

typedef __attribute__((ext_vector_type(16))) _Float16 v16h;
typedef __attribute__((ext_vector_type(8)))  _Float16 v8h;
typedef __attribute__((ext_vector_type(16))) __bf16   v16bf;
typedef __attribute__((ext_vector_type(8)))  __bf16   v8bf;
typedef __attribute__((ext_vector_type(8)))  float    v8f;

// float -> bf16 (RNE)
__device__ __forceinline__ __bf16 f2bf(float f) {
    unsigned int u = __builtin_bit_cast(unsigned int, f);
    u += 0x7fffu + ((u >> 16) & 1u);
    unsigned short h = (unsigned short)(u >> 16);
    return __builtin_bit_cast(__bf16, h);
}

__global__ void k_zero(float* __restrict__ p, int n)
{
    const int i = blockIdx.x * blockDim.x + threadIdx.x;
    if (i < n) p[i] = 0.f;
}

// ---------------------------------------------------------------------------
// Expert kernel: block = 8 waves, handles 16 adjacent columns j0..j0+15 of U
// (one expert) over an n-partition. Each wave-step processes ONE row n for
// all 16 columns: the WMMA A-matrix rows are the 16 columns' h1 vectors.
//   h1 = relu(U[n,j]*w1+b1); h2 = relu(h1@w2 + b2); v = h2.w3
//   Vfull[j] += sum_n v   (atomic across n-partitions; /N and +b3 in k_wvec)
// n is wave-uniform -> coalesced U loads and zero per-lane masking.
// 2 K-chunks x 4 N-tiles = 8 WMMAs per (wave, n).
// ---------------------------------------------------------------------------
__global__ void __launch_bounds__(256)
k_expert(const float* __restrict__ U,
         const float* __restrict__ w1, const float* __restrict__ b1,
         const float* __restrict__ w2, const float* __restrict__ b2,
         const float* __restrict__ w3,
         float* __restrict__ Vfull)
{
    // w2 stored transposed: w2sT[out_ch][in_ch]; rows padded to 72 hw = 144B
    // (multiple of 16B so every fragment base is b128-aligned).
    __shared__ __attribute__((aligned(16))) _Float16 w2sT[EH][72];
    __shared__ float w1s[EH], b1s[EH], b2s[EH], w3s[EH];
    __shared__ float partial[8][16];

    const int j0 = blockIdx.x * 16;    // 16-column group (within one expert)
    const int e  = j0 / KPE;           // expert id (400 % 16 == 0)
    const int tid  = threadIdx.x;
    const int wave = tid >> 5;
    const int lane = tid & 31;
    const int l15  = lane & 15;
    const int hi   = lane >> 4;

    for (int i = tid; i < EH * EH; i += 256) {
        const int r = i >> 6, c = i & 63;          // coalesced read of w2[r][c]
        w2sT[c][r] = (_Float16)w2[e * EH * EH + r * EH + c];
    }
    if (tid < EH) {
        w1s[tid] = w1[e * EH + tid];
        b1s[tid] = b1[e * EH + tid];
        b2s[tid] = b2[e * EH + tid];
        w3s[tid] = w3[e * EH + tid];
    }
    __syncthreads();

    // B fragments (w2), 2 K-chunks x 4 N-tiles. Fragment elem[2v+p] =
    // w2[q*32 + 16*hi + (0..15)][t*16 + l15] -> contiguous run in w2sT.
    v16h bfr[2][4];
#pragma unroll
    for (int q = 0; q < 2; ++q)
#pragma unroll
        for (int t = 0; t < 4; ++t) {
            const _Float16* p = &w2sT[t * 16 + l15][q * 32 + 16 * hi];
            v8h lo = *(const v8h*)p;
            v8h hh = *(const v8h*)(p + 8);
            bfr[q][t] = __builtin_shufflevector(lo, hh, 0, 1, 2, 3, 4, 5, 6, 7,
                                                8, 9, 10, 11, 12, 13, 14, 15);
        }

    // Hoist lane-constant layer params into registers.
    float w1r[2][16], b1r[2][16];
#pragma unroll
    for (int v = 0; v < 8; ++v)
#pragma unroll
        for (int p = 0; p < 2; ++p) {
            const int kidx = 2 * v + p + 16 * hi;
            w1r[0][2 * v + p] = w1s[kidx];       b1r[0][2 * v + p] = b1s[kidx];
            w1r[1][2 * v + p] = w1s[32 + kidx];  b1r[1][2 * v + p] = b1s[32 + kidx];
        }
    float bbr[4], w3r[4];
#pragma unroll
    for (int t = 0; t < 4; ++t) {
        bbr[t] = b2s[t * 16 + l15];
        w3r[t] = w3s[t * 16 + l15];
    }

    // acc[v]: partial sum for column j0 + v + 8*hi (summed over n, channels)
    float acc[8];
#pragma unroll
    for (int v = 0; v < 8; ++v) acc[v] = 0.f;

    const int nstart = blockIdx.y * (NN / NPART);
    const int nend   = nstart + (NN / NPART);

    for (int n = nstart + wave; n < nend; n += 8) {   // wave-uniform bound
        const float u = U[(size_t)n * NN + j0 + l15]; // coalesced 64B run

        // A fragments: row m = column index; lane supplies row l15's h1.
        v16h a0, a1;
#pragma unroll
        for (int i = 0; i < 16; ++i) {
            const float h0 = fmaf(u, w1r[0][i], b1r[0][i]);
            const float h1 = fmaf(u, w1r[1][i], b1r[1][i]);
            a0[i] = (_Float16)fmaxf(h0, 0.f);
            a1[i] = (_Float16)fmaxf(h1, 0.f);
        }

#pragma unroll
        for (int t = 0; t < 4; ++t) {
            v8f d;
#pragma unroll
            for (int v = 0; v < 8; ++v) d[v] = bbr[t];   // fold +b2 into C
            d = __builtin_amdgcn_wmma_f32_16x16x32_f16(false, a0, false, bfr[0][t],
                                                       (short)0, d, false, false);
            d = __builtin_amdgcn_wmma_f32_16x16x32_f16(false, a1, false, bfr[1][t],
                                                       (short)0, d, false, false);
            const float ww3 = w3r[t];
#pragma unroll
            for (int v = 0; v < 8; ++v)
                acc[v] += fmaxf(d[v], 0.f) * ww3;        // relu . w3
        }
    }

    // Reduce over the 16 lanes of each half-wave (same column set per half).
#pragma unroll
    for (int off = 8; off; off >>= 1)
#pragma unroll
        for (int v = 0; v < 8; ++v)
            acc[v] += __shfl_xor(acc[v], off, 32);
    if (l15 == 0) {
#pragma unroll
        for (int v = 0; v < 8; ++v)
            partial[wave][v + 8 * hi] = acc[v];
    }
    __syncthreads();
    if (tid < 16) {
        float s = 0.f;
#pragma unroll
        for (int w = 0; w < 8; ++w) s += partial[w][tid];
        atomicAdd(&Vfull[j0 + tid], s);
    }
}

// ---------------------------------------------------------------------------
// Gating: stats = mean(La per group); gate = softmax(relu(stats@g1+b1)@g2+b2)
// ---------------------------------------------------------------------------
__global__ void k_gate(const float* __restrict__ La,
                       const float* __restrict__ g1w, const float* __restrict__ g1b,
                       const float* __restrict__ g2w, const float* __restrict__ g2b,
                       float* __restrict__ gate)
{
    __shared__ float stats[NE], a[NE];
    const int t = threadIdx.x;
    if (t < NE) {
        float s = 0.f;
        for (int k = 0; k < KPE; ++k) s += La[t * KPE + k];
        stats[t] = s / (float)KPE;
    }
    __syncthreads();
    if (t < NE) {
        float z = g1b[t];
        for (int i = 0; i < NE; ++i) z += stats[i] * g1w[i * NE + t];
        a[t] = z > 0.f ? z : 0.f;
    }
    __syncthreads();
    if (t == 0) {
        float z[NE], mx = -1e30f, sum = 0.f;
        for (int o = 0; o < NE; ++o) {
            float v = g2b[o];
            for (int i = 0; i < NE; ++i) v += a[i] * g2w[i * NE + o];
            z[o] = v; mx = fmaxf(mx, v);
        }
        for (int o = 0; o < NE; ++o) { z[o] = __expf(z[o] - mx); sum += z[o]; }
        for (int o = 0; o < NE; ++o) gate[o] = z[o] / sum;
    }
}

// wvec[j] = gate[e] * ( Vfull[j]/N + b3[e] )
__global__ void k_wvec(const float* __restrict__ Vfull, const float* __restrict__ gate,
                       const float* __restrict__ b3, float* __restrict__ wvec)
{
    const int i = blockIdx.x * blockDim.x + threadIdx.x;
    if (i < NN) {
        const int e = i / KPE;
        wvec[i] = gate[e] * (Vfull[i] * (1.f / (float)NN) + b3[e]);
    }
}

// ---------------------------------------------------------------------------
// Generic WMMA GEMM: C[MxNn] = op(A) * B (+bias), f32 in/out, bf16 compute.
//   transA: A stored [Kk x M] (lda=M) and read transposed (coalesced path).
//   scale : optional per-k scale on A (implements U*w column scaling).
// Block = 128 thr (4 waves), 64x64 tile, K-step 32. Each wave owns a 16-row
// strip with 4 accumulators -> 4 WMMAs per barrier pair. B staged transposed
// in LDS so both fragments load as 2x ds_load_b128 per lane.
// ---------------------------------------------------------------------------
__global__ void __launch_bounds__(128)
k_gemm(float* __restrict__ C, const float* __restrict__ A, const float* __restrict__ B,
       int M, int Nn, int Kk, int lda, int ldb, int transA,
       const float* __restrict__ scale, const float* __restrict__ bias)
{
    __shared__ __attribute__((aligned(16))) __bf16 As[64][40];   // [m][k], 80B rows
    __shared__ __attribute__((aligned(16))) __bf16 BsT[64][40];  // [n][k]

    const int tid  = threadIdx.x;
    const int wave = tid >> 5;
    const int lane = tid & 31;
    const int l15  = lane & 15;
    const int hi   = lane >> 4;
    const int m0 = blockIdx.x * 64, n0 = blockIdx.y * 64;
    const int wrow = wave * 16;

    v8f d[4];
#pragma unroll
    for (int t = 0; t < 4; ++t) d[t] = (v8f){};

    for (int k0 = 0; k0 < Kk; k0 += 32) {
        // ---- stage A (64m x 32k) ----
        if (!transA) {
            for (int i = tid; i < 64 * 32; i += 128) {
                const int r = i >> 5, c = i & 31;          // coalesced over k
                const int mm = m0 + r, kk = k0 + c;
                float av = 0.f;
                if (mm < M && kk < Kk) {
                    av = A[(size_t)mm * lda + kk];
                    if (scale) av *= scale[kk];
                }
                As[r][c] = f2bf(av);
            }
        } else {
            for (int i = tid; i < 64 * 32; i += 128) {
                const int c = i >> 6, r = i & 63;          // coalesced over m
                const int mm = m0 + r, kk = k0 + c;
                float av = 0.f;
                if (mm < M && kk < Kk) {
                    av = A[(size_t)kk * lda + mm];
                    if (scale) av *= scale[kk];
                }
                As[r][c] = f2bf(av);
            }
        }
        // ---- stage B (32k x 64n), transposed into BsT[n][k] ----
        for (int i = tid; i < 32 * 64; i += 128) {
            const int r = i >> 6, c = i & 63;              // coalesced over n
            const int kk = k0 + r, nn = n0 + c;
            float bv = 0.f;
            if (kk < Kk && nn < Nn) bv = B[(size_t)kk * ldb + nn];
            BsT[c][r] = f2bf(bv);
        }
        if (k0 + 32 < Kk)
            __builtin_prefetch(&B[(size_t)(k0 + 32) * ldb + n0], 0, 1);
        __syncthreads();

        // A fragment: elem[2v+p] -> k = 8*hi + (0..7) | 16+8*hi + (0..7)
        {
            const __bf16* pa = &As[wrow + l15][0];
            v8bf alo = *(const v8bf*)(pa + 8 * hi);
            v8bf ahi = *(const v8bf*)(pa + 16 + 8 * hi);
            v16bf a = __builtin_shufflevector(alo, ahi, 0, 1, 2, 3, 4, 5, 6, 7,
                                              8, 9, 10, 11, 12, 13, 14, 15);
#pragma unroll
            for (int t = 0; t < 4; ++t) {
                const __bf16* pb = &BsT[t * 16 + l15][8 * hi];
                v8bf blo = *(const v8bf*)pb;
                v8bf bhi = *(const v8bf*)(pb + 16);
                v16bf b = __builtin_shufflevector(blo, bhi, 0, 1, 2, 3, 4, 5, 6, 7,
                                                  8, 9, 10, 11, 12, 13, 14, 15);
                d[t] = __builtin_amdgcn_wmma_f32_16x16x32_bf16(false, a, false, b,
                                                               (short)0, d[t],
                                                               false, false);
            }
        }
        __syncthreads();
    }

#pragma unroll
    for (int t = 0; t < 4; ++t) {
        const int n = n0 + t * 16 + l15;
        const float bs = (bias && n < Nn) ? bias[n] : 0.f;
#pragma unroll
        for (int v = 0; v < 8; ++v) {
            const int m = m0 + wrow + v + 8 * hi;
            if (m < M && n < Nn) C[(size_t)m * Nn + n] = d[t][v] + bs;
        }
    }
}

// ---------------------------------------------------------------------------
// BatchNorm batch stats (training mode, biased var): one thread per channel,
// coalesced column walk.
// ---------------------------------------------------------------------------
__global__ void k_bn_stats(const float* __restrict__ X, int n, int c,
                           float* __restrict__ mean, float* __restrict__ rstd)
{
    const int ch = blockIdx.x * blockDim.x + threadIdx.x;
    if (ch >= c) return;
    float s = 0.f, s2 = 0.f;
    for (int i = 0; i < n; ++i) {
        const float x = X[(size_t)i * c + ch];
        s += x; s2 += x * x;
    }
    const float m = s / (float)n;
    const float v = s2 / (float)n - m * m;
    mean[ch] = m;
    rstd[ch] = rsqrtf(v + BN_EPS);
}

// T = relu( relu(bn1(H)) + L )
__global__ void k_ew01(const float* __restrict__ H, const float* __restrict__ L,
                       const float* __restrict__ mean, const float* __restrict__ rstd,
                       const float* __restrict__ g, const float* __restrict__ b,
                       float* __restrict__ T, int total)
{
    const int i = blockIdx.x * blockDim.x + threadIdx.x;
    if (i >= total) return;
    const int ch = i & (HIDD - 1);
    float h = (H[i] - mean[ch]) * rstd[ch] * g[ch] + b[ch];
    h = fmaxf(h, 0.f) + L[i];
    T[i] = fmaxf(h, 0.f);
}

// X = bn2(T)
__global__ void k_bnapply(const float* __restrict__ T,
                          const float* __restrict__ mean, const float* __restrict__ rstd,
                          const float* __restrict__ g, const float* __restrict__ b,
                          float* __restrict__ X, int total)
{
    const int i = blockIdx.x * blockDim.x + threadIdx.x;
    if (i >= total) return;
    const int ch = i & (HIDD - 1);
    X[i] = (T[i] - mean[ch]) * rstd[ch] * g[ch] + b[ch];
}

// h = relu(bn1(H))
__global__ void k_ewh(const float* __restrict__ H,
                      const float* __restrict__ mean, const float* __restrict__ rstd,
                      const float* __restrict__ g, const float* __restrict__ b,
                      float* __restrict__ Out, int total)
{
    const int i = blockIdx.x * blockDim.x + threadIdx.x;
    if (i >= total) return;
    const int ch = i & (HIDD - 1);
    Out[i] = fmaxf((H[i] - mean[ch]) * rstd[ch] * g[ch] + b[ch], 0.f);
}

// S = sigmoid(S) + L   (in place on S)
__global__ void k_ewsig(float* __restrict__ S, const float* __restrict__ L, int total)
{
    const int i = blockIdx.x * blockDim.x + threadIdx.x;
    if (i >= total) return;
    S[i] = 1.f / (1.f + __expf(-S[i])) + L[i];
}

// row-wise log_softmax over 64 columns; one wave per row, 2 cols per lane
__global__ void k_logsoftmax(const float* __restrict__ P, float* __restrict__ Out)
{
    const int row = blockIdx.x;
    const int lane = threadIdx.x;           // 0..31
    const float x0 = P[(size_t)row * OUTC + lane];
    const float x1 = P[(size_t)row * OUTC + lane + 32];
    float mx = fmaxf(x0, x1);
#pragma unroll
    for (int off = 16; off; off >>= 1) mx = fmaxf(mx, __shfl_xor(mx, off, 32));
    float s = __expf(x0 - mx) + __expf(x1 - mx);
#pragma unroll
    for (int off = 16; off; off >>= 1) s += __shfl_xor(s, off, 32);
    const float lse = mx + __logf(s);
    Out[(size_t)row * OUTC + lane]      = x0 - lse;
    Out[(size_t)row * OUTC + lane + 32] = x1 - lse;
}

// ---------------------------------------------------------------------------
// Host driver. Input order assumes recursive insertion-order flattening of
// setup_inputs(): 0:X 1:La 2:U, then morgan{0,1,2} each
// [w1,b1,w2,b2,w3,b3,g1w,g1b,g2w,g2b,Ww,Wb], then line{0..2}{w,b},
// bn1_{0..2}{g,b}, bn2_{0,1}{g,b}, mlpw, mlpb.
// ---------------------------------------------------------------------------
extern "C" void kernel_launch(void* const* d_in, const int* in_sizes, int n_in,
                              void* d_out, int out_size, void* d_ws, size_t ws_size,
                              hipStream_t stream)
{
    (void)in_sizes; (void)n_in; (void)out_size; (void)ws_size;
    const float* X0 = (const float*)d_in[0];
    const float* La = (const float*)d_in[1];
    const float* U  = (const float*)d_in[2];
    auto P = [&](int i) { return (const float*)d_in[i]; };

    float* w = (float*)d_ws;
    float* Vfull = w;                 // 2048
    float* gate  = w + 2048;          // 8
    float* wvec  = w + 2064;          // 2048
    float* mean1 = w + 4112;          // 256
    float* rstd1 = w + 4368;          // 256
    float* mean2 = w + 4624;          // 256
    float* rstd2 = w + 4880;          // 256
    float* Z  = w + 8192;             // N*HID
    float* Y  = Z  + NN * HIDD;
    float* H  = Y  + NN * HIDD;
    float* T  = H  + NN * HIDD;
    float* L  = T  + NN * HIDD;
    float* S  = L  + NN * HIDD;       // N*OUTC
    float* XA = S  + NN * OUTC;
    float* XB = XA + NN * HIDD;

    const float* Xcur = X0;
    int Cin = IN_C;
    const dim3 blk128(128);
    const int gx = (NN + 63) / 64;    // 32

    for (int l = 0; l < 3; ++l) {
        const int base = 3 + l * 12;
        const float *w1 = P(base + 0), *b1 = P(base + 1), *w2 = P(base + 2);
        const float *b2 = P(base + 3), *w3 = P(base + 4), *b3 = P(base + 5);
        const float *g1w = P(base + 6), *g1b = P(base + 7);
        const float *g2w = P(base + 8), *g2b = P(base + 9);
        const float *Ww = P(base + 10), *Wb = P(base + 11);

        k_zero<<<(NN + 255) / 256, 256, 0, stream>>>(Vfull, NN);
        k_expert<<<dim3(NN / 16, NPART), 256, 0, stream>>>(U, w1, b1, w2, b2, w3,
                                                           Vfull);
        k_gate<<<1, 32, 0, stream>>>(La, g1w, g1b, g2w, g2b, gate);
        k_wvec<<<(NN + 255) / 256, 256, 0, stream>>>(Vfull, gate, b3, wvec);

        // Z = U^T Xcur              [N x Cin]
        k_gemm<<<dim3(gx, (Cin + 63) / 64), blk128, 0, stream>>>(
            Z, U, Xcur, NN, Cin, NN, NN, Cin, 1, nullptr, nullptr);
        // Y = (U * w) Z             [N x Cin]
        k_gemm<<<dim3(gx, (Cin + 63) / 64), blk128, 0, stream>>>(
            Y, U, Z, NN, Cin, NN, NN, Cin, 0, wvec, nullptr);
        // H = Y Ww + Wb             [N x HID]
        k_gemm<<<dim3(gx, HIDD / 64), blk128, 0, stream>>>(
            H, Y, Ww, NN, HIDD, Cin, Cin, HIDD, 0, nullptr, Wb);

        k_bn_stats<<<1, HIDD, 0, stream>>>(H, NN, HIDD, mean1, rstd1);

        const float* bg1 = P(45 + 2 * l);
        const float* bb1 = P(46 + 2 * l);

        if (l < 2) {
            const float* lw = P(39 + 2 * l);
            const float* lb = P(40 + 2 * l);
            // L = Xcur lineW + lineB
            k_gemm<<<dim3(gx, HIDD / 64), blk128, 0, stream>>>(
                L, Xcur, lw, NN, HIDD, Cin, Cin, HIDD, 0, nullptr, lb);
            k_ew01<<<(NN * HIDD + 255) / 256, 256, 0, stream>>>(
                H, L, mean1, rstd1, bg1, bb1, T, NN * HIDD);
            k_bn_stats<<<1, HIDD, 0, stream>>>(T, NN, HIDD, mean2, rstd2);
            const float* bg2 = P(51 + 2 * l);
            const float* bb2 = P(52 + 2 * l);
            float* Xn = (l == 0) ? XA : XB;
            k_bnapply<<<(NN * HIDD + 255) / 256, 256, 0, stream>>>(
                T, mean2, rstd2, bg2, bb2, Xn, NN * HIDD);
            Xcur = Xn;
            Cin = HIDD;
        } else {
            // h = relu(bn1(H)) -> T
            k_ewh<<<(NN * HIDD + 255) / 256, 256, 0, stream>>>(
                H, mean1, rstd1, bg1, bb1, T, NN * HIDD);
            const float* mlpw = P(55); const float* mlpb = P(56);
            // S = h mlpw + mlpb      [N x 64]
            k_gemm<<<dim3(gx, 1), blk128, 0, stream>>>(
                S, T, mlpw, NN, OUTC, HIDD, HIDD, OUTC, 0, nullptr, mlpb);
            const float* l2w = P(43); const float* l2b = P(44);
            // L = Xcur line2w + line2b ([N x 64])
            k_gemm<<<dim3(gx, 1), blk128, 0, stream>>>(
                L, Xcur, l2w, NN, OUTC, HIDD, HIDD, OUTC, 0, nullptr, l2b);
            k_ewsig<<<(NN * OUTC + 255) / 256, 256, 0, stream>>>(S, L, NN * OUTC);
            k_logsoftmax<<<NN, 32, 0, stream>>>(S, (float*)d_out);
        }
    }
}
